// TransformerClassifier_48498770707323
// MI455X (gfx1250) — matrix operand order repaired and checked by reading, not compiled, not run on hardware
//
#include <hip/hip_runtime.h>
#include <hip/hip_bf16.h>

typedef __attribute__((ext_vector_type(16))) __bf16 v16bf;
typedef __attribute__((ext_vector_type(8)))  __bf16 v8bf;
typedef __attribute__((ext_vector_type(8)))  float  v8f;

#define DIM     768
#define N_NODES 50000
#define N_EDGES 8000
#define NT      4        // N tiles (16 cols each) per wave in the GEMM

// ---------------------------------------------------------------------------
// fp32 -> bf16 elementwise (GEMM A operand)
// ---------------------------------------------------------------------------
__global__ void k_f32_to_bf16(const float* __restrict__ in,
                              __bf16* __restrict__ out, long n) {
    long i = (long)blockIdx.x * blockDim.x + threadIdx.x;
    if (i < n) out[i] = (__bf16)in[i];
}

// Transpose+convert W[k][n] -> Wt[n][k] bf16 (B fragment contiguous in K)
__global__ void k_transpose_w_bf16(const float* __restrict__ W,
                                   __bf16* __restrict__ Wt) {
    int n = blockIdx.x;
    for (int k = threadIdx.x; k < DIM; k += blockDim.x)
        Wt[(size_t)n * DIM + k] = (__bf16)W[(size_t)k * DIM + n];
}

// ---------------------------------------------------------------------------
// GEMM: C = A(bf16) @ W + b.  Wave computes a 16x64 strip: A fragment loaded
// once per K-step, reused across 4 WMMAs with 4 independent acc chains.
// grid = (M/16, 3), block = 128 (4 waves); wave w -> tn group by*4+w (0..11).
// ---------------------------------------------------------------------------
__global__ void k_gemm_wmma_bf16(const __bf16* __restrict__ A,
                                 const __bf16* __restrict__ Wt,
                                 const float* __restrict__ bias,
                                 float* __restrict__ C) {
    const int lane = threadIdx.x & 31;
    const int wave = threadIdx.x >> 5;            // 0..3
    const int tm   = blockIdx.x;                  // 0..3124
    const int tng  = blockIdx.y * 4 + wave;       // 0..11
    const int half = lane >> 4;
    const int l16  = lane & 15;

    // A frag: row = tm*16+l16; element j -> K = (j/8)*16 + half*8 + j%8
    const __bf16* Ap = A + (size_t)(tm * 16 + l16) * DIM + half * 8;
    // B frags: col = tn*16+l16; element j -> K = (j/8)*16 + half*8 + j%8
    const __bf16* Bp[NT];
#pragma unroll
    for (int t = 0; t < NT; ++t)
        Bp[t] = Wt + (size_t)((tng * NT + t) * 16 + l16) * DIM + half * 8;

    v8f acc[NT];
#pragma unroll
    for (int t = 0; t < NT; ++t) acc[t] = (v8f){};

    for (int k0 = 0; k0 < DIM; k0 += 32) {
        v8bf a0 = *(const v8bf*)(Ap + k0);
        v8bf a1 = *(const v8bf*)(Ap + k0 + 16);
        v16bf a;
#pragma unroll
        for (int j = 0; j < 8; ++j) { a[j] = a0[j]; a[j + 8] = a1[j]; }
#pragma unroll
        for (int t = 0; t < NT; ++t) {
            v8bf b0 = *(const v8bf*)(Bp[t] + k0);
            v8bf b1 = *(const v8bf*)(Bp[t] + k0 + 16);
            v16bf b;
#pragma unroll
            for (int j = 0; j < 8; ++j) { b[j] = b0[j]; b[j + 8] = b1[j]; }
            acc[t] = __builtin_amdgcn_wmma_f32_16x16x32_bf16(
                false, a, false, b, (short)0, acc[t], false, false);
        }
    }

    // D layout: VGPR r -> row = r + 8*half, col = l16 (per tile)
#pragma unroll
    for (int t = 0; t < NT; ++t) {
        const int col = (tng * NT + t) * 16 + l16;
        const float bb = bias[col];
#pragma unroll
        for (int r = 0; r < 8; ++r) {
            const int row = tm * 16 + r + half * 8;
            C[(size_t)row * DIM + col] = acc[t][r] + bb;
        }
    }
}

// ---------------------------------------------------------------------------
// CSR construction (built once per launch, shared by both layers)
// ---------------------------------------------------------------------------
__global__ void k_zero_i32(int* __restrict__ p, int n) {
    int i = blockIdx.x * blockDim.x + threadIdx.x;
    if (i < n) p[i] = 0;
}

__global__ void k_degree(const int* __restrict__ V, const int* __restrict__ E,
                         int* __restrict__ vdeg, int* __restrict__ edeg, int n) {
    int i = blockIdx.x * blockDim.x + threadIdx.x;
    if (i < n) {
        atomicAdd(&edeg[E[i]], 1);
        atomicAdd(&vdeg[V[i]], 1);
    }
}

// Single-block exclusive scan, n <= 1024*chunk; writes out[0..n], out[n]=total
__global__ void k_scan_excl(const int* __restrict__ in, int* __restrict__ out,
                            int n) {
    __shared__ int part[1024];
    const int T = 1024;
    const int t = threadIdx.x;
    const int chunk = (n + T - 1) / T;
    int lo = t * chunk;      if (lo > n) lo = n;
    int hi = lo + chunk;     if (hi > n) hi = n;
    int s = 0;
    for (int i = lo; i < hi; ++i) s += in[i];
    part[t] = s;
    __syncthreads();
    for (int off = 1; off < T; off <<= 1) {
        int v = (t >= off) ? part[t - off] : 0;
        __syncthreads();
        part[t] += v;
        __syncthreads();
    }
    int base = (t == 0) ? 0 : part[t - 1];
    for (int i = lo; i < hi; ++i) { out[i] = base; base += in[i]; }
    if (t == T - 1) out[n] = part[T - 1];
}

__global__ void k_copy_i32(const int* __restrict__ in, int* __restrict__ out,
                           int n) {
    int i = blockIdx.x * blockDim.x + threadIdx.x;
    if (i < n) out[i] = in[i];
}

// Fill both adjacency lists: elist[e's slots] = v, vlist[v's slots] = e
__global__ void k_fill(const int* __restrict__ V, const int* __restrict__ E,
                       int* __restrict__ ecur, int* __restrict__ vcur,
                       int* __restrict__ elist, int* __restrict__ vlist, int n) {
    int i = blockIdx.x * blockDim.x + threadIdx.x;
    if (i < n) {
        int v = V[i], e = E[i];
        int p = atomicAdd(&ecur[e], 1);
        elist[p] = v;
        int q = atomicAdd(&vcur[v], 1);
        vlist[q] = e;
    }
}

// ---------------------------------------------------------------------------
// Xe[e] = mean_{v in e} Y[v].  One block per edge; pure gather (no atomics).
// Y (153.6MB) and Xe (24.6MB) are L2-resident -> L2-bandwidth bound.
// ---------------------------------------------------------------------------
__global__ void k_edge_aggregate(const float* __restrict__ Y,
                                 const int* __restrict__ eoff,
                                 const int* __restrict__ elist,
                                 float* __restrict__ Xe) {
    const int e  = blockIdx.x;
    const int lo = eoff[e], hi = eoff[e + 1];
    const float inv = 1.0f / fmaxf((float)(hi - lo), 1.0f);
    const int t = threadIdx.x;
    float a0 = 0.0f, a1 = 0.0f, a2 = 0.0f;
    for (int p = lo; p < hi; ++p) {
        const float* row = Y + (size_t)elist[p] * DIM;
        a0 += row[t];
        a1 += row[t + 256];
        a2 += row[t + 512];
    }
    float* dst = Xe + (size_t)e * DIM;
    dst[t]       = a0 * inv;
    dst[t + 256] = a1 * inv;
    dst[t + 512] = a2 * inv;
}

// ---------------------------------------------------------------------------
// Fused: Y[v] += sum_{e ∋ v} Xe[e];  then row L2-normalize; optional ReLU.
// One block (256 thr) per node; 3 columns per thread; single pass over Y.
// ---------------------------------------------------------------------------
__global__ void k_node_update(float* __restrict__ Y,
                              const int* __restrict__ voff,
                              const int* __restrict__ vlist,
                              const float* __restrict__ Xe, int relu) {
    __shared__ float red[8];
    const int v  = blockIdx.x;
    const int lo = voff[v], hi = voff[v + 1];
    const int t = threadIdx.x;
    float* row = Y + (size_t)v * DIM;
    float x0 = row[t], x1 = row[t + 256], x2 = row[t + 512];
    for (int p = lo; p < hi; ++p) {
        const float* er = Xe + (size_t)vlist[p] * DIM;
        x0 += er[t];
        x1 += er[t + 256];
        x2 += er[t + 512];
    }
    float s = x0 * x0 + x1 * x1 + x2 * x2;
#pragma unroll
    for (int off = 16; off > 0; off >>= 1)
        s += __shfl_down(s, off, 32);
    const int lane = t & 31, wave = t >> 5;
    if (lane == 0) red[wave] = s;
    __syncthreads();
    if (t == 0) {
        float tot = 0.0f;
#pragma unroll
        for (int w = 0; w < 8; ++w) tot += red[w];
        red[0] = 1.0f / fmaxf(sqrtf(tot), 1e-12f);
    }
    __syncthreads();
    const float scale = red[0];
    x0 *= scale; x1 *= scale; x2 *= scale;
    if (relu) {
        x0 = fmaxf(x0, 0.0f); x1 = fmaxf(x1, 0.0f); x2 = fmaxf(x2, 0.0f);
    }
    row[t] = x0; row[t + 256] = x1; row[t + 512] = x2;
}

// ---------------------------------------------------------------------------
static inline size_t alignup(size_t x) { return (x + 255) & ~(size_t)255; }

extern "C" void kernel_launch(void* const* d_in, const int* in_sizes, int n_in,
                              void* d_out, int out_size, void* d_ws, size_t ws_size,
                              hipStream_t stream) {
    const float* x  = (const float*)d_in[0];
    const float* W1 = (const float*)d_in[1];
    const float* b1 = (const float*)d_in[2];
    const float* W2 = (const float*)d_in[3];
    const float* b2 = (const float*)d_in[4];
    const int*   V  = (const int*)d_in[5];
    const int*   E  = (const int*)d_in[6];
    float* out = (float*)d_out;
    const int n_inc = in_sizes[5];

    // Workspace carve-up
    char* ws = (char*)d_ws;
    float*  Y1  = (float*)ws;   ws += alignup((size_t)N_NODES * DIM * 4);
    __bf16* Xbf = (__bf16*)ws;  ws += alignup((size_t)N_NODES * DIM * 2);
    __bf16* Wbf = (__bf16*)ws;  ws += alignup((size_t)DIM * DIM * 2);
    float*  Xe  = (float*)ws;   ws += alignup((size_t)N_EDGES * DIM * 4);
    int* edeg  = (int*)ws;      ws += alignup((size_t)N_EDGES * 4);
    int* vdeg  = (int*)ws;      ws += alignup((size_t)N_NODES * 4);
    int* eoff  = (int*)ws;      ws += alignup((size_t)(N_EDGES + 1) * 4);
    int* voff  = (int*)ws;      ws += alignup((size_t)(N_NODES + 1) * 4);
    int* ecur  = (int*)ws;      ws += alignup((size_t)N_EDGES * 4);
    int* vcur  = (int*)ws;      ws += alignup((size_t)N_NODES * 4);
    int* elist = (int*)ws;      ws += alignup((size_t)n_inc * 4);
    int* vlist = (int*)ws;

    // ---- CSR build (once; reused by both layers) ----
    k_zero_i32<<<(N_EDGES + 255) / 256, 256, 0, stream>>>(edeg, N_EDGES);
    k_zero_i32<<<(N_NODES + 255) / 256, 256, 0, stream>>>(vdeg, N_NODES);
    k_degree<<<(n_inc + 255) / 256, 256, 0, stream>>>(V, E, vdeg, edeg, n_inc);
    k_scan_excl<<<1, 1024, 0, stream>>>(edeg, eoff, N_EDGES);
    k_scan_excl<<<1, 1024, 0, stream>>>(vdeg, voff, N_NODES);
    k_copy_i32<<<(N_EDGES + 255) / 256, 256, 0, stream>>>(eoff, ecur, N_EDGES);
    k_copy_i32<<<(N_NODES + 255) / 256, 256, 0, stream>>>(voff, vcur, N_NODES);
    k_fill<<<(n_inc + 255) / 256, 256, 0, stream>>>(V, E, ecur, vcur,
                                                    elist, vlist, n_inc);

    const long nX = (long)N_NODES * DIM;
    const dim3 gemm_grid(N_NODES / 16, 3);   // 3 * 4 waves * 4 tiles = 48 N-tiles

    auto layer = [&](const float* Xin, const float* W, const float* b,
                     float* Yout, int relu) {
        k_transpose_w_bf16<<<DIM, 256, 0, stream>>>(W, Wbf);
        k_f32_to_bf16<<<(int)((nX + 255) / 256), 256, 0, stream>>>(Xin, Xbf, nX);
        k_gemm_wmma_bf16<<<gemm_grid, 128, 0, stream>>>(Xbf, Wbf, b, Yout);
        k_edge_aggregate<<<N_EDGES, 256, 0, stream>>>(Yout, eoff, elist, Xe);
        k_node_update<<<N_NODES, 256, 0, stream>>>(Yout, voff, vlist, Xe, relu);
    };

    layer(x,  W1, b1, Y1,  1);  // layer 1 + ReLU
    layer(Y1, W2, b2, out, 0);  // layer 2 -> d_out
}
